// NVFP4Linear_68951404970059
// MI455X (gfx1250) — 8-bit matrix operand order repaired and checked by reading, compiled, not run on hardware
//
#include <hip/hip_runtime.h>

typedef __attribute__((ext_vector_type(16))) int   v16i;
typedef __attribute__((ext_vector_type(8)))  float v8f;
typedef __attribute__((ext_vector_type(4)))  int   v4i;

#define M_DIM 512
#define N_DIM 4096
#define K_DIM 4096

#define BM 64
#define BN 128
#define BK 128
#define KTILES (K_DIM / BK)
#define LSTR 144
#define A_BYTES (BM * LSTR)
#define B_BYTES (BN * LSTR)
#define BUF_BYTES (A_BYTES + B_BYTES)

#if __has_builtin(__builtin_amdgcn_global_load_async_to_lds_b128)
#define HAVE_ASYNC_LDS 1
#else
#define HAVE_ASYNC_LDS 0
#endif

__device__ __forceinline__ void async_b128(const void* g, void* l) {
#if HAVE_ASYNC_LDS
    __builtin_amdgcn_global_load_async_to_lds_b128(
        (__attribute__((address_space(1))) v4i*)g,
        (__attribute__((address_space(3))) v4i*)l,
        0, 0);
#else
    *(uint4*)l = *(const uint4*)g;
#endif
}

template <int N>
__device__ __forceinline__ void wait_async() {
#if HAVE_ASYNC_LDS
#if __has_builtin(__builtin_amdgcn_s_wait_asynccnt)
    __builtin_amdgcn_s_wait_asynccnt(N);
#else
    asm volatile("s_wait_asynccnt %0" ::"i"(N) : "memory");
#endif
#endif
}

__device__ __forceinline__ void stage_tile(unsigned char* smem,
                                           const unsigned char* __restrict__ xq,
                                           const unsigned char* __restrict__ wq,
                                           int mBase, int nBase, int k,
                                           int t, int buf) {
    unsigned char* sA = smem + buf * BUF_BYTES;
    unsigned char* sB = sA + A_BYTES;
    const int row8 = t >> 3;
    const int col  = (t & 7) * 16;
    #pragma unroll
    for (int i = 0; i < BM / 32; ++i) {
        const int r = i * 32 + row8;
        async_b128(xq + (size_t)(mBase + r) * K_DIM + k + col,
                   sA + r * LSTR + col);
    }
    #pragma unroll
    for (int i = 0; i < BN / 32; ++i) {
        const int r = i * 32 + row8;
        async_b128(wq + (size_t)(nBase + r) * K_DIM + k + col,
                   sB + r * LSTR + col);
    }
}

__global__ void __launch_bounds__(256)
nvfp4_amax_kernel(const float4* __restrict__ in, int n4, unsigned* __restrict__ amax_bits) {
    float m = 0.0f;
    for (int i = blockIdx.x * blockDim.x + threadIdx.x; i < n4;
         i += gridDim.x * blockDim.x) {
        float4 v = in[i];
        m = fmaxf(m, fabsf(v.x));
        m = fmaxf(m, fabsf(v.y));
        m = fmaxf(m, fabsf(v.z));
        m = fmaxf(m, fabsf(v.w));
    }
    #pragma unroll
    for (int off = 16; off > 0; off >>= 1)
        m = fmaxf(m, __shfl_down(m, off, 32));
    __shared__ float smax[8];
    const int lane = threadIdx.x & 31;
    const int wave = threadIdx.x >> 5;
    if (lane == 0) smax[wave] = m;
    __syncthreads();
    if (wave == 0) {
        m = (lane < 8) ? smax[lane] : 0.0f;
        #pragma unroll
        for (int off = 16; off > 0; off >>= 1)
            m = fmaxf(m, __shfl_down(m, off, 32));
        if (lane == 0)
            atomicMax(amax_bits, __float_as_uint(m));
    }
}

__device__ __forceinline__ unsigned nvfp4_q1(float v, float scale) {
    float s = v * scale;
    unsigned sign = (__float_as_uint(s) >> 31) << 7;
    float a = fminf(fabsf(s), 6.0f);
    unsigned code;
    if      (a <= 0.25f) code = 0x00;
    else if (a <= 0.75f) code = 0x30;
    else if (a <= 1.25f) code = 0x38;
    else if (a <= 1.75f) code = 0x3C;
    else if (a <= 2.50f) code = 0x40;
    else if (a <= 3.50f) code = 0x44;
    else if (a <= 5.00f) code = 0x48;
    else                 code = 0x4C;
    return code | sign;
}

__global__ void __launch_bounds__(256)
nvfp4_quant_kernel(const float4* __restrict__ in, unsigned* __restrict__ out,
                   int n4, const unsigned* __restrict__ amax_bits) {
    const float amax  = fmaxf(__uint_as_float(*amax_bits), 1e-12f);
    const float scale = 6.0f / amax;
    for (int i = blockIdx.x * blockDim.x + threadIdx.x; i < n4;
         i += gridDim.x * blockDim.x) {
        float4 v = in[i];
        unsigned b =  nvfp4_q1(v.x, scale)
                   | (nvfp4_q1(v.y, scale) << 8)
                   | (nvfp4_q1(v.z, scale) << 16)
                   | (nvfp4_q1(v.w, scale) << 24);
        out[i] = b;
    }
}

__global__ void __launch_bounds__(256)
nvfp4_gemm_fp8_kernel(const unsigned char* __restrict__ xq,
                      const unsigned char* __restrict__ wq,
                      float* __restrict__ out,
                      const unsigned* __restrict__ amax_x_bits,
                      const unsigned* __restrict__ amax_w_bits) {
    __shared__ __align__(16) unsigned char smem[2 * BUF_BYTES];

    const int t      = threadIdx.x;
    const int lane   = t & 31;
    const int wave   = t >> 5;
    const int laneLo = lane & 15;
    const bool hi    = lane >= 16;
    const int mw     = wave >> 1;
    const int nw     = wave & 1;

    const int mBase = blockIdx.y * BM;
    const int nBase = blockIdx.x * BN;

    stage_tile(smem, xq, wq, mBase, nBase, 0,  t, 0);
    stage_tile(smem, xq, wq, mBase, nBase, BK, t, 1);

    v8f acc[4] = {};

    for (int kt = 0; kt < KTILES; ++kt) {
        const int cur = kt & 1;
        if (kt < KTILES - 1) wait_async<6>();
        else                 wait_async<0>();
        __syncthreads();

        const unsigned char* sA = smem + cur * BUF_BYTES;
        const unsigned char* sB = sA + A_BYTES;

        v16i a;
        const unsigned char* aRow = sA + (mw * 16 + laneLo) * LSTR;
        #pragma unroll
        for (int q = 0; q < 8; ++q) {
            const int kb = (q >> 1) * 32 + (q & 1) * 8 + (hi ? 16 : 0);
            const uint2 d = *(const uint2*)(aRow + kb);
            a[2 * q]     = (int)d.x;
            a[2 * q + 1] = (int)d.y;
        }

        #pragma unroll
        for (int s = 0; s < 4; ++s) {
            const unsigned char* bRow = sB + (nw * 64 + s * 16 + laneLo) * LSTR;
            v16i b;
            #pragma unroll
            for (int g = 0; g < 4; ++g) {
                const int kb = g * 32 + (hi ? 16 : 0);
                const uint4 d = *(const uint4*)(bRow + kb);
                b[4 * g]     = (int)d.x;
                b[4 * g + 1] = (int)d.y;
                b[4 * g + 2] = (int)d.z;
                b[4 * g + 3] = (int)d.w;
            }
            acc[s] = __builtin_amdgcn_wmma_f32_16x16x128_fp8_fp8(
                a, b, (short)0, acc[s], false, false);
        }

        __syncthreads();
        if (kt + 2 < KTILES)
            stage_tile(smem, xq, wq, mBase, nBase, (kt + 2) * BK, t, cur);
    }

    const float ax  = fmaxf(__uint_as_float(*amax_x_bits), 1e-12f);
    const float aw  = fmaxf(__uint_as_float(*amax_w_bits), 1e-12f);
    const float mul = (ax * (1.0f / 6.0f)) * (aw * (1.0f / 6.0f));

    #pragma unroll
    for (int s = 0; s < 4; ++s) {
        const int n = nBase + nw * 64 + s * 16 + laneLo;
        #pragma unroll
        for (int r = 0; r < 8; ++r) {
            const int m = mBase + mw * 16 + r + (hi ? 8 : 0);
            out[(size_t)m * N_DIM + n] = acc[s][r] * mul;
        }
    }
}

extern "C" void kernel_launch(void* const* d_in, const int* in_sizes, int n_in,
                              void* d_out, int out_size, void* d_ws, size_t ws_size,
                              hipStream_t stream) {
    const float* x = (const float*)d_in[0];
    const float* w = (const float*)d_in[1];
    float* out     = (float*)d_out;

    unsigned char* ws = (unsigned char*)d_ws;
    unsigned char* xq = ws;
    unsigned char* wq = ws + (size_t)M_DIM * K_DIM;
    unsigned* amax    = (unsigned*)(ws + (size_t)M_DIM * K_DIM
                                       + (size_t)N_DIM * K_DIM);

    const int nx4 = (M_DIM * K_DIM) / 4;
    const int nw4 = (N_DIM * K_DIM) / 4;

    (void)hipMemsetAsync(amax, 0, 2 * sizeof(unsigned), stream);

    nvfp4_amax_kernel<<<512, 256, 0, stream>>>((const float4*)x, nx4, amax + 0);
    nvfp4_amax_kernel<<<1024, 256, 0, stream>>>((const float4*)w, nw4, amax + 1);

    nvfp4_quant_kernel<<<512, 256, 0, stream>>>((const float4*)x, (unsigned*)xq, nx4, amax + 0);
    nvfp4_quant_kernel<<<2048, 256, 0, stream>>>((const float4*)w, (unsigned*)wq, nw4, amax + 1);

    dim3 grid(N_DIM / BN, M_DIM / BM);
    nvfp4_gemm_fp8_kernel<<<grid, 256, 0, stream>>>(xq, wq, out, amax + 0, amax + 1);
}
